// location_head_8589934592114
// MI455X (gfx1250) — matrix operand order repaired and checked by reading, not compiled, not run on hardware
//
#include <hip/hip_runtime.h>
#include <hip/hip_bf16.h>
#include <math.h>

typedef _Float16 h16;
typedef __attribute__((ext_vector_type(16))) _Float16 v16h;
typedef __attribute__((ext_vector_type(8)))  float    v8f;

#define EPS 1e-5f

__device__ inline void wave_lds_sync() {
    // same-wave LDS RAW across lanes: wait for DS ops + compiler barrier
    asm volatile("s_wait_dscnt 0" ::: "memory");
}

__device__ inline float sigmoidf_(float x) { return 1.0f / (1.0f + __expf(-x)); }

// ---------------------------------------------------------------------------
// Encoder: per-node conv blocks + maxpool + station MLP -> X0 f16 [N,192]
// One wave per node, 4 waves (128 threads) per block. LDS 4*3216 f32 = 51.5KB.
// ---------------------------------------------------------------------------
__global__ void __launch_bounds__(128)
k_encode(const float* __restrict__ x, const float* __restrict__ sloc,
         const float* __restrict__ c1w, const float* __restrict__ c1b,
         const float* __restrict__ bn1g, const float* __restrict__ bn1b,
         const float* __restrict__ bn1m, const float* __restrict__ bn1v,
         const float* __restrict__ c2w, const float* __restrict__ c2b,
         const float* __restrict__ bn2g, const float* __restrict__ bn2b,
         const float* __restrict__ bn2m, const float* __restrict__ bn2v,
         const float* __restrict__ g1w, const float* __restrict__ g1b,
         const float* __restrict__ gbn1g, const float* __restrict__ gbn1b,
         const float* __restrict__ gbn1m, const float* __restrict__ gbn1v,
         const float* __restrict__ g2w, const float* __restrict__ g2b,
         const float* __restrict__ gbn2g, const float* __restrict__ gbn2b,
         const float* __restrict__ gbn2m, const float* __restrict__ gbn2v,
         const float* __restrict__ s1w, const float* __restrict__ s1b,
         const float* __restrict__ s2w, const float* __restrict__ s2b,
         h16* __restrict__ X0, int n_nodes)
{
    __shared__ float lds[4][3216];
    const int wave = threadIdx.x >> 5, lane = threadIdx.x & 31;
    const int n = blockIdx.x * 4 + wave;
    if (n >= n_nodes) return;
    float* buf = lds[wave];

    for (int i = lane; i < 144; i += 32) buf[i] = x[(size_t)n * 144 + i];
    wave_lds_sync();

    float* A = buf + 144;   // [32,48]
    float* B = buf + 1680;  // [32,48]

    // gd1.c1: 3->32 k7 same, + BN + ReLU  (lane = out channel)
    {
        const int oc = lane;
        float w[21];
        #pragma unroll
        for (int t = 0; t < 21; t++) w[t] = c1w[oc * 21 + t];
        const float bb = c1b[oc];
        const float sc = bn1g[oc] * rsqrtf(bn1v[oc] + EPS);
        const float sh = bn1b[oc] - bn1m[oc] * sc;
        for (int p = 0; p < 48; p++) {
            float s = bb;
            #pragma unroll
            for (int ic = 0; ic < 3; ic++)
                #pragma unroll
                for (int t = 0; t < 7; t++) {
                    int ip = p + t - 3;
                    if (ip >= 0 && ip < 48) s += w[ic * 7 + t] * buf[ic * 48 + ip];
                }
            float v = s * sc + sh;
            A[oc * 48 + p] = v > 0.0f ? v : 0.0f;
        }
    }
    wave_lds_sync();

    // gd1.c2: 32->32 k7 same, + BN + ReLU
    {
        const int oc = lane;
        float acc[48];
        const float bb = c2b[oc];
        #pragma unroll
        for (int p = 0; p < 48; p++) acc[p] = bb;
        for (int ic = 0; ic < 32; ic++) {
            float w[7];
            #pragma unroll
            for (int t = 0; t < 7; t++) w[t] = c2w[oc * 224 + ic * 7 + t];
            float ar[48];
            #pragma unroll
            for (int p = 0; p < 48; p++) ar[p] = A[ic * 48 + p];
            #pragma unroll
            for (int p = 0; p < 48; p++)
                #pragma unroll
                for (int t = 0; t < 7; t++) {
                    int ip = p + t - 3;
                    if (ip >= 0 && ip < 48) acc[p] += w[t] * ar[ip];
                }
        }
        const float sc = bn2g[oc] * rsqrtf(bn2v[oc] + EPS);
        const float sh = bn2b[oc] - bn2m[oc] * sc;
        #pragma unroll
        for (int p = 0; p < 48; p++) {
            float v = acc[p] * sc + sh;
            B[oc * 48 + p] = v > 0.0f ? v : 0.0f;
        }
    }
    wave_lds_sync();

    // maxpool(5, stride 2, pad 2): [32,48] -> P [32,24]
    float* P = buf;
    {
        const int oc = lane;
        for (int p = 0; p < 24; p++) {
            float m = -3.0e38f;
            #pragma unroll
            for (int j = 0; j < 5; j++) {
                int ip = 2 * p - 2 + j;
                if (ip >= 0 && ip < 48) { float v = B[oc * 48 + ip]; m = v > m ? v : m; }
            }
            P[oc * 24 + p] = m;
        }
    }
    wave_lds_sync();

    // gd2.c1: 32->4 k7 + BN + ReLU -> Q [4,24]
    float* Q = buf + 768;
    for (int idx = lane; idx < 96; idx += 32) {
        int oc = idx / 24, p = idx % 24;
        float s = g1b[oc];
        for (int ic = 0; ic < 32; ic++)
            #pragma unroll
            for (int t = 0; t < 7; t++) {
                int ip = p + t - 3;
                if (ip >= 0 && ip < 24) s += g1w[oc * 224 + ic * 7 + t] * P[ic * 24 + ip];
            }
        float sc = gbn1g[oc] * rsqrtf(gbn1v[oc] + EPS);
        float v = s * sc + (gbn1b[oc] - gbn1m[oc] * sc);
        Q[idx] = v > 0.0f ? v : 0.0f;
    }
    wave_lds_sync();

    // gd2.c2: 4->4 k7 + BN + ReLU -> R [4,24] == flattened h96
    float* R = buf + 864;
    for (int idx = lane; idx < 96; idx += 32) {
        int oc = idx / 24, p = idx % 24;
        float s = g2b[oc];
        #pragma unroll
        for (int ic = 0; ic < 4; ic++)
            #pragma unroll
            for (int t = 0; t < 7; t++) {
                int ip = p + t - 3;
                if (ip >= 0 && ip < 24) s += g2w[oc * 28 + ic * 7 + t] * Q[ic * 24 + ip];
            }
        float sc = gbn2g[oc] * rsqrtf(gbn2v[oc] + EPS);
        float v = s * sc + (gbn2b[oc] - gbn2m[oc] * sc);
        R[idx] = v > 0.0f ? v : 0.0f;
    }
    wave_lds_sync();

    // station MLP: 3 -> 48 (ReLU) -> 96
    const float sl0 = sloc[(size_t)n * 3 + 0], sl1 = sloc[(size_t)n * 3 + 1],
                sl2 = sloc[(size_t)n * 3 + 2];
    float* S = buf + 960;
    for (int j = lane; j < 48; j += 32) {
        float v = s1b[j] + s1w[j * 3 + 0] * sl0 + s1w[j * 3 + 1] * sl1 + s1w[j * 3 + 2] * sl2;
        S[j] = v > 0.0f ? v : 0.0f;
    }
    wave_lds_sync();
    for (int j = lane; j < 96; j += 32) {
        float v = s2b[j];
        #pragma unroll 8
        for (int i = 0; i < 48; i++) v += s2w[j * 48 + i] * S[i];
        X0[(size_t)n * 192 + 96 + j] = (h16)v;
        X0[(size_t)n * 192 + j] = (h16)R[j];
    }
}

// ---------------------------------------------------------------------------
// WMMA GEMM, NT 16x16 output tiles per wave (A fragment reused NT times):
// Y[M, Ncols] = act(Xf16[M,ldx] @ W16[Ncols,ldw]^T + bias)
// ---------------------------------------------------------------------------
template <int NT>
__global__ void __launch_bounds__(256)
k_gemm(const h16* __restrict__ X, int ldx,
       const h16* __restrict__ W, int ldw,
       const float* __restrict__ bias,
       float* __restrict__ Yf, int ldyf,
       h16* __restrict__ Yh, int ldyh,
       int M, int Ncols, int K, int relu)
{
    const int wid  = (blockIdx.x * blockDim.x + threadIdx.x) >> 5;
    const int lane = threadIdx.x & 31;
    const int grpN = Ncols / (16 * NT);
    const int tiles = (M >> 4) * grpN;
    if (wid >= tiles) return;                           // wave-uniform
    const int tm = wid / grpN, tg = wid % grpN;
    const int half = lane >> 4, l16 = lane & 15;
    const int arow = tm * 16 + l16;
    const h16* __restrict__ xr = X + (size_t)arow * ldx;

    const h16* wr[NT];
    int bcol[NT];
    #pragma unroll
    for (int j = 0; j < NT; j++) {
        bcol[j] = (tg * NT + j) * 16 + l16;
        wr[j] = W + (size_t)bcol[j] * ldw;
    }

    v8f acc[NT];
    #pragma unroll
    for (int j = 0; j < NT; j++) acc[j] = (v8f){};

    for (int kk = 0; kk < K; kk += 32) {
        if (kk + 32 < K) __builtin_prefetch(xr + kk + 32, 0, 0);
        union { v16h v; unsigned u[8]; } a;
        #pragma unroll
        for (int r = 0; r < 8; r++) {
            // A 16x32 f16 layout: K = (r<4?0:16) + 8*half + 2*(r&3)
            int ka = kk + ((r & 4) ? 16 : 0) + 8 * half + 2 * (r & 3);
            a.u[r] = *(const unsigned*)(xr + ka);
        }
        #pragma unroll
        for (int j = 0; j < NT; j++) {
            union { v16h v; unsigned u[8]; } b;
            #pragma unroll
            for (int r = 0; r < 8; r++) {
                // B 32x16 f16 layout: col = lane%16, K = (r<4?0:16) + 8*half + 2*(r&3)
                int kb = kk + ((r & 4) ? 16 : 0) + 8 * half + 2 * (r & 3);
                b.u[r] = *(const unsigned*)(wr[j] + kb);
            }
            acc[j] = __builtin_amdgcn_wmma_f32_16x16x32_f16(false, a.v, false, b.v,
                                                            (short)0, acc[j], false, false);
        }
    }
    #pragma unroll
    for (int j = 0; j < NT; j++) {
        const float bv = bias[bcol[j]];
        #pragma unroll
        for (int r = 0; r < 8; r++) {
            int row = tm * 16 + r + 8 * half;
            float y = acc[j][r] + bv;
            if (relu) y = fmaxf(y, 0.0f);
            if (Yf) Yf[(size_t)row * ldyf + bcol[j]] = y;
            if (Yh) Yh[(size_t)row * ldyh + bcol[j]] = (h16)y;
        }
    }
}

// ---------------------------------------------------------------------------
// Weight packing: f32 [O,K] -> f16 [colOff+O rows, Kpad] zero-padded; bias f32
// ---------------------------------------------------------------------------
__global__ void k_pack_w(const float* __restrict__ w, const float* __restrict__ b,
                         h16* __restrict__ wout, float* __restrict__ bout,
                         int O, int K, int Kpad, int colOff)
{
    int i = blockIdx.x * blockDim.x + threadIdx.x;
    if (i < O * Kpad) {
        int o = i / Kpad, k = i % Kpad;
        wout[(size_t)(colOff + o) * Kpad + k] = (k < K) ? (h16)w[o * K + k] : (h16)0.0f;
    }
    if (i < O) bout[colOff + i] = b[i];
}

__global__ void k_fill_h(h16* p, int n, float v) {
    int i = blockIdx.x * blockDim.x + threadIdx.x;
    if (i < n) p[i] = (h16)v;
}
__global__ void k_fill_f(float* p, int n, float v) {
    int i = blockIdx.x * blockDim.x + threadIdx.x;
    if (i < n) p[i] = v;
}
__global__ void k_fill_i(int* p, int n, int v) {
    int i = blockIdx.x * blockDim.x + threadIdx.x;
    if (i < n) p[i] = v;
}

// ---------------------------------------------------------------------------
// CSR build (incoming edges per dst): count -> exclusive scan -> scatter
// ---------------------------------------------------------------------------
__global__ void k_count(const int* __restrict__ dst, int* __restrict__ cnt, int E) {
    int e = blockIdx.x * blockDim.x + threadIdx.x;
    if (e < E) atomicAdd(&cnt[dst[e]], 1);
}

// single-block exclusive scan of cnt[0..n) -> rowptr[0..n], rowptr[n] = total
__global__ void __launch_bounds__(1024)
k_scan(const int* __restrict__ cnt, int* __restrict__ rowptr, int n)
{
    __shared__ int sh[1024];
    __shared__ int carry;
    if (threadIdx.x == 0) carry = 0;
    __syncthreads();
    for (int base = 0; base < n; base += 1024) {
        int i = base + (int)threadIdx.x;
        int v = (i < n) ? cnt[i] : 0;
        sh[threadIdx.x] = v;
        __syncthreads();
        for (int o = 1; o < 1024; o <<= 1) {
            int t = (threadIdx.x >= (unsigned)o) ? sh[threadIdx.x - o] : 0;
            __syncthreads();
            sh[threadIdx.x] += t;
            __syncthreads();
        }
        if (i < n) rowptr[i] = sh[threadIdx.x] - v + carry;   // exclusive
        __syncthreads();
        if (threadIdx.x == 1023) carry += sh[1023];
        __syncthreads();
    }
    if (threadIdx.x == 0) rowptr[n] = carry;
}

__global__ void k_copy_i(const int* __restrict__ a, int* __restrict__ b, int n) {
    int i = blockIdx.x * blockDim.x + threadIdx.x;
    if (i < n) b[i] = a[i];
}

__global__ void k_scatter(const int* __restrict__ dst, int* __restrict__ cursor,
                          int* __restrict__ eids, int E) {
    int e = blockIdx.x * blockDim.x + threadIdx.x;
    if (e >= E) return;
    int p = atomicAdd(&cursor[dst[e]], 1);
    eids[p] = e;
}

// ---------------------------------------------------------------------------
// TransformerConv softmax+aggregate: one wave per dst node, zero atomics.
// Lanes cover channels d = lane, lane+32 (D <= 48). Per incoming edge, the
// logit dot-product is reduced with a wave butterfly; alpha-weighted v[src]
// accumulates in registers; result is added onto the skip columns in place.
// ---------------------------------------------------------------------------
__global__ void __launch_bounds__(256)
k_attn(const int* __restrict__ rowptr, const int* __restrict__ eids,
       const int* __restrict__ src, float* __restrict__ Q, int ld,
       int qoff, int koff, int voff, int aggoff, int D, float scale, int N)
{
    const int wid = (blockIdx.x * blockDim.x + threadIdx.x) >> 5;
    const int lane = threadIdx.x & 31;
    if (wid >= N) return;                               // wave-uniform
    const int beg = rowptr[wid], end = rowptr[wid + 1];
    if (beg == end) return;                             // isolated: agg = skip

    const bool a0 = lane < D, a1 = lane + 32 < D;
    const float q0 = a0 ? Q[(size_t)wid * ld + qoff + lane] : 0.0f;
    const float q1 = a1 ? Q[(size_t)wid * ld + qoff + lane + 32] : 0.0f;

    // pass 1: running max of logits
    float mx = -3.0e38f;
    for (int i = beg; i < end; i++) {
        const int s = src[eids[i]];
        float p = q0 * (a0 ? Q[(size_t)s * ld + koff + lane] : 0.0f)
                + q1 * (a1 ? Q[(size_t)s * ld + koff + lane + 32] : 0.0f);
        #pragma unroll
        for (int o = 16; o > 0; o >>= 1) p += __shfl_xor(p, o, 32);
        mx = fmaxf(mx, p * scale);
    }
    // pass 2: exp-sum + weighted aggregation of v[src]
    float ssum = 0.0f, acc0 = 0.0f, acc1 = 0.0f;
    for (int i = beg; i < end; i++) {
        const int s = src[eids[i]];
        float p = q0 * (a0 ? Q[(size_t)s * ld + koff + lane] : 0.0f)
                + q1 * (a1 ? Q[(size_t)s * ld + koff + lane + 32] : 0.0f);
        #pragma unroll
        for (int o = 16; o > 0; o >>= 1) p += __shfl_xor(p, o, 32);
        const float w = __expf(p * scale - mx);         // identical on all lanes
        ssum += w;
        acc0 += w * (a0 ? Q[(size_t)s * ld + voff + lane] : 0.0f);
        acc1 += w * (a1 ? Q[(size_t)s * ld + voff + lane + 32] : 0.0f);
    }
    const float inv = 1.0f / ssum;
    if (a0) Q[(size_t)wid * ld + aggoff + lane] += acc0 * inv;
    if (a1) Q[(size_t)wid * ld + aggoff + lane + 32] += acc1 * inv;
}

// tc1 epilogue: pack x_temp into padded f16 [N,64] and write f32 x_temp output
__global__ void k_finish_tc1(const float* __restrict__ Q, h16* __restrict__ xt,
                             float* __restrict__ out_xtemp, int N)
{
    int i = blockIdx.x * blockDim.x + threadIdx.x;
    if (i >= N * 64) return;
    int n = i / 64, d = i % 64;
    float v = (d < 48) ? Q[(size_t)n * 192 + 144 + d] : 0.0f;
    xt[i] = (h16)v;
    if (d < 48) out_xtemp[(size_t)n * 48 + d] = v;
}

__global__ void k_finish_tc2(const float* __restrict__ Q, h16* __restrict__ h2, int N)
{
    int i = blockIdx.x * blockDim.x + threadIdx.x;
    if (i >= N * 32) return;
    int n = i / 32, d = i % 32;
    h2[i] = (h16)((d < 12) ? Q[(size_t)n * 48 + 36 + d] : 0.0f);
}

__global__ void k_offset_out(const float* __restrict__ Q, float* __restrict__ out, int N)
{
    int n = blockIdx.x * blockDim.x + threadIdx.x;
    if (n < N) out[n] = sigmoidf_(Q[(size_t)n * 16 + 3]);
}

// batch is repeat(arange(G), 50): nodes of graph g are rows [g*50, g*50+50)
__global__ void k_depth_out(const float* __restrict__ Q, float* __restrict__ out, int G)
{
    int g = blockIdx.x * blockDim.x + threadIdx.x;
    if (g >= G) return;
    float s = 0.0f;
    for (int i = 0; i < 50; i++) s += Q[(size_t)(g * 50 + i) * 16 + 3];
    out[g] = sigmoidf_(s / 50.0f);
}

// ---------------------------------------------------------------------------
extern "C" void kernel_launch(void* const* d_in, const int* in_sizes, int n_in,
                              void* d_out, int out_size, void* d_ws, size_t ws_size,
                              hipStream_t stream)
{
    const int N = in_sizes[1] / 3;          // station_loc [N,3]
    const int E = in_sizes[3] / 2;          // edge_index [2,E]
    const int G = N / 50;

    const float* x    = (const float*)d_in[0];
    const float* sloc = (const float*)d_in[1];
    const int*   ei   = (const int*)d_in[3];
    const int *src = ei, *dst = ei + E;

    #define FP(i) ((const float*)d_in[(i)])

    size_t off = 0;
    auto alloc = [&](size_t bytes) -> void* {
        off = (off + 255) & ~(size_t)255;
        void* p = (char*)d_ws + off;
        off += bytes;
        return p;
    };
    h16*   X0    = (h16*)  alloc((size_t)N * 192 * 2);
    h16*   H1    = (h16*)  alloc((size_t)N * 96 * 2);
    h16*   Hf    = (h16*)  alloc((size_t)N * 96 * 2);
    float* QK1   = (float*)alloc((size_t)N * 192 * 4);
    h16*   xt    = (h16*)  alloc((size_t)N * 64 * 2);
    float* QK2   = (float*)alloc((size_t)N * 48 * 4);
    h16*   h2    = (h16*)  alloc((size_t)N * 32 * 2);
    float* QK3   = (float*)alloc((size_t)N * 16 * 4);
    float* QK4   = (float*)alloc((size_t)N * 16 * 4);
    int*   cnt   = (int*)  alloc((size_t)N * 4);
    int*   rowptr= (int*)  alloc((size_t)(N + 1) * 4);
    int*   cursor= (int*)  alloc((size_t)N * 4);
    int*   eids  = (int*)  alloc((size_t)E * 4);
    h16*   Wt1   = (h16*)  alloc(96 * 192 * 2);  float* Bt1 = (float*)alloc(96 * 4);
    h16*   Wt2   = (h16*)  alloc(96 * 96 * 2);   float* Bt2 = (float*)alloc(96 * 4);
    h16*   W1    = (h16*)  alloc(192 * 96 * 2);  float* B1  = (float*)alloc(192 * 4);
    h16*   W2    = (h16*)  alloc(48 * 64 * 2);   float* B2  = (float*)alloc(48 * 4);
    h16*   W3    = (h16*)  alloc(16 * 32 * 2);   float* B3  = (float*)alloc(16 * 4);
    h16*   W4    = (h16*)  alloc(16 * 32 * 2);   float* B4  = (float*)alloc(16 * 4);

    float* out_offset = (float*)d_out;          // [N]
    float* out_depth  = out_offset + N;         // [G]
    float* out_xtemp  = out_depth + G;          // [N,48]

    // 1) encoder + CSR build (independent streams of work, same queue)
    k_encode<<<(N + 3) / 4, 128, 0, stream>>>(
        x, sloc,
        FP(4), FP(5), FP(6), FP(7), FP(8), FP(9),
        FP(10), FP(11), FP(12), FP(13), FP(14), FP(15),
        FP(16), FP(17), FP(18), FP(19), FP(20), FP(21),
        FP(22), FP(23), FP(24), FP(25), FP(26), FP(27),
        FP(28), FP(29), FP(30), FP(31), X0, N);

    k_fill_i<<<(N + 255) / 256, 256, 0, stream>>>(cnt, N, 0);
    k_count<<<(E + 255) / 256, 256, 0, stream>>>(dst, cnt, E);
    k_scan<<<1, 1024, 0, stream>>>(cnt, rowptr, N);
    k_copy_i<<<(N + 255) / 256, 256, 0, stream>>>(rowptr, cursor, N);
    k_scatter<<<(E + 255) / 256, 256, 0, stream>>>(dst, cursor, eids, E);

    // 2) pack weights (pad K to 32-multiples, cols to 16; zero tiny head blocks)
    auto pack = [&](int wi, int bi, h16* wo, float* bo, int O, int K, int Kpad, int colOff) {
        int tot = O * Kpad;
        k_pack_w<<<(tot + 255) / 256, 256, 0, stream>>>(FP(wi), FP(bi), wo, bo, O, K, Kpad, colOff);
    };
    k_fill_h<<<2, 256, 0, stream>>>(W3, 16 * 32, 0.0f);
    k_fill_h<<<2, 256, 0, stream>>>(W4, 16 * 32, 0.0f);
    k_fill_f<<<1, 64, 0, stream>>>(B3, 16, 0.0f);
    k_fill_f<<<1, 64, 0, stream>>>(B4, 16, 0.0f);
    pack(32, 33, Wt1, Bt1, 96, 192, 192, 0);                     // tmlp.fc1
    pack(34, 35, Wt2, Bt2, 96, 96, 96, 0);                       // tmlp.fc2
    pack(36, 37, W1, B1, 48, 96, 96, 0);                         // tc1 q
    pack(38, 39, W1, B1, 48, 96, 96, 48);                        // tc1 k
    pack(40, 41, W1, B1, 48, 96, 96, 96);                        // tc1 v
    pack(42, 43, W1, B1, 48, 96, 96, 144);                       // tc1 skip
    pack(44, 45, W2, B2, 12, 48, 64, 0);                         // tc2 q
    pack(46, 47, W2, B2, 12, 48, 64, 12);
    pack(48, 49, W2, B2, 12, 48, 64, 24);
    pack(50, 51, W2, B2, 12, 48, 64, 36);
    pack(52, 53, W3, B3, 1, 12, 32, 0);                          // tc3 q,k,v,skip
    pack(54, 55, W3, B3, 1, 12, 32, 1);
    pack(56, 57, W3, B3, 1, 12, 32, 2);
    pack(58, 59, W3, B3, 1, 12, 32, 3);
    pack(60, 61, W4, B4, 1, 12, 32, 0);                          // tcd
    pack(62, 63, W4, B4, 1, 12, 32, 1);
    pack(64, 65, W4, B4, 1, 12, 32, 2);
    pack(66, 67, W4, B4, 1, 12, 32, 3);

    // 3) WMMA GEMM chain (NT tiles per wave for A-fragment reuse)
    auto gemm = [&](const h16* X, int ldx, const h16* W, int ldw, const float* b,
                    float* Yf, int ldyf, h16* Yh, int ldyh, int Ncols, int K, int relu) {
        int nt = (Ncols % 64 == 0) ? 4 : (Ncols % 48 == 0) ? 3 : (Ncols % 32 == 0) ? 2 : 1;
        int tiles = (N >> 4) * (Ncols / (16 * nt));
        int blocks = (tiles * 32 + 255) / 256;
        switch (nt) {
        case 4: k_gemm<4><<<blocks, 256, 0, stream>>>(X, ldx, W, ldw, b, Yf, ldyf, Yh, ldyh, N, Ncols, K, relu); break;
        case 3: k_gemm<3><<<blocks, 256, 0, stream>>>(X, ldx, W, ldw, b, Yf, ldyf, Yh, ldyh, N, Ncols, K, relu); break;
        case 2: k_gemm<2><<<blocks, 256, 0, stream>>>(X, ldx, W, ldw, b, Yf, ldyf, Yh, ldyh, N, Ncols, K, relu); break;
        default: k_gemm<1><<<blocks, 256, 0, stream>>>(X, ldx, W, ldw, b, Yf, ldyf, Yh, ldyh, N, Ncols, K, relu); break;
        }
    };
    gemm(X0, 192, Wt1, 192, Bt1, nullptr, 0, H1, 96, 96, 192, 1);   // tmlp fc1 + relu
    gemm(H1, 96, Wt2, 96, Bt2, nullptr, 0, Hf, 96, 96, 96, 0);      // tmlp fc2
    gemm(Hf, 96, W1, 96, B1, QK1, 192, nullptr, 0, 192, 96, 0);     // tc1 q|k|v|skip

    // 4) attention layers: one wave per dst node, atomic-free
    auto attn = [&](float* Q, int ld, int qo, int ko, int vo, int ao, int D, float scale) {
        int blocks = (N * 32 + 255) / 256;
        k_attn<<<blocks, 256, 0, stream>>>(rowptr, eids, src, Q, ld, qo, ko, vo, ao, D, scale, N);
    };
    attn(QK1, 192, 0, 48, 96, 144, 48, rsqrtf(48.0f));              // tc1
    k_finish_tc1<<<((N * 64) + 255) / 256, 256, 0, stream>>>(QK1, xt, out_xtemp, N);

    gemm(xt, 64, W2, 64, B2, QK2, 48, nullptr, 0, 48, 64, 0);       // tc2 projections
    attn(QK2, 48, 0, 12, 24, 36, 12, rsqrtf(12.0f));
    k_finish_tc2<<<((N * 32) + 255) / 256, 256, 0, stream>>>(QK2, h2, N);

    gemm(h2, 32, W3, 32, B3, QK3, 16, nullptr, 0, 16, 32, 0);       // tc3 projections
    gemm(h2, 32, W4, 32, B4, QK4, 16, nullptr, 0, 16, 32, 0);       // tcd projections
    attn(QK3, 16, 0, 1, 2, 3, 1, 1.0f);                             // tc3
    k_offset_out<<<(N + 255) / 256, 256, 0, stream>>>(QK3, out_offset, N);
    attn(QK4, 16, 0, 1, 2, 3, 1, 1.0f);                             // tcd
    k_depth_out<<<(G + 255) / 256, 256, 0, stream>>>(QK4, out_depth, G);

    #undef FP
    (void)ws_size; (void)n_in; (void)out_size;
}
